// HfMistral4Attention_17085379904042
// MI455X (gfx1250) — compile-verified, matrix operand order checked by reading, not run on hardware
//
#include <hip/hip_runtime.h>
#include <hip/hip_bf16.h>

// ---------------------------------------------------------------------------
// MLA attention (DeepSeek-style) for MI455X / gfx1250.
// All matmuls on v_wmma_f32_16x16x32_bf16; all operands pre-converted to bf16
// in global memory so LDS staging is pure 16B copies, done with
// GLOBAL_LOAD_ASYNC_TO_LDS_B128 (ASYNCcnt) when the toolchain exposes it.
// ---------------------------------------------------------------------------

typedef __attribute__((ext_vector_type(8)))  unsigned short ushort8;
typedef __attribute__((ext_vector_type(16))) unsigned short ushort16;
typedef __attribute__((ext_vector_type(16))) __bf16        v16bf;
typedef __attribute__((ext_vector_type(8)))  float         v8f;

#if defined(__has_builtin)
#  if __has_builtin(__builtin_amdgcn_global_load_async_to_lds_b128)
#    define HAVE_ASYNC_LDS 1
#  endif
#  if __has_builtin(__builtin_amdgcn_s_wait_asynccnt)
#    define HAVE_WAIT_ASYNC 1
#  endif
#endif

// Builtin prototype (from hipcc diagnostic): first param is
// 'int __vector_size__(16) __device__ *'  == int4 addrspace(1)*,
// second (by the *1/*3 convention) int4 addrspace(3)*.
typedef int v4i_t __attribute__((vector_size(16)));
typedef __attribute__((address_space(1))) v4i_t gv4i;
typedef __attribute__((address_space(3))) v4i_t lv4i;

__device__ __forceinline__ unsigned short f2bf(float f) {
  unsigned int u = __float_as_uint(f);
  unsigned int r = (u + 0x7FFFu + ((u >> 16) & 1u)) >> 16;   // RNE
  return (unsigned short)r;
}
__device__ __forceinline__ float bf2f(unsigned short u) {
  return __uint_as_float((unsigned int)u << 16);
}

// 16-byte global -> LDS copy. Async (no VGPR round trip, ASYNCcnt) if the
// builtin exists; else a synchronous vector copy.
__device__ __forceinline__ void cp16_g2l(const unsigned short* g, unsigned short* l) {
#if defined(HAVE_ASYNC_LDS)
  __builtin_amdgcn_global_load_async_to_lds_b128(
      (gv4i*)(unsigned long long)(uintptr_t)g,
      (lv4i*)(unsigned int)(uintptr_t)l,     // AS(3) ptrs are 32-bit offsets
      0, 0);
#else
  *(ushort8*)l = *(const ushort8*)g;
#endif
}

__device__ __forceinline__ void wait_cp() {
#if defined(HAVE_ASYNC_LDS)
#  if defined(HAVE_WAIT_ASYNC)
  __builtin_amdgcn_s_wait_asynccnt(0);
#  else
  asm volatile("s_wait_asynccnt 0" ::: "memory");
#  endif
#endif
}

// Load a 16(M/N) x 32(K) bf16 WMMA fragment from a row-major tile.
// 'p' points at this lane's row (32 K-values, 16B aligned). ISA 16-bit layout:
//   lanes 0-15 : K = 0..7 and 16..23 ; lanes 16-31: K = 8..15 and 24..31
__device__ __forceinline__ v16bf load_frag(const unsigned short* p, int half) {
  ushort8 lo = *(const ushort8*)(p + half * 8);
  ushort8 hi = *(const ushort8*)(p + 16 + half * 8);
  ushort16 u = __builtin_shufflevector(lo, hi, 0, 1, 2, 3, 4, 5, 6, 7,
                                       8, 9, 10, 11, 12, 13, 14, 15);
  return __builtin_bit_cast(v16bf, u);
}

__device__ __forceinline__ v8f wmma_bf16(v16bf a, v16bf b, v8f c) {
  return __builtin_amdgcn_wmma_f32_16x16x32_bf16(
      false, a, false, b, (short)0, c, false, false);
}

// ---------------------------------------------------------------------------
// f32 -> bf16 tensor conversion (done once per tensor, not once per tile use)
// ---------------------------------------------------------------------------
__global__ __launch_bounds__(256) void cvt_bf16_k(const float* __restrict__ s,
                                                  unsigned short* __restrict__ d,
                                                  int n) {
  for (int i = blockIdx.x * 256 + threadIdx.x; i < n; i += gridDim.x * 256)
    d[i] = f2bf(s[i]);
}

// ---------------------------------------------------------------------------
// RoPE cache: scale==1 -> inv_freq[j] = 10000^(-j/32), mscale == 1.
// ---------------------------------------------------------------------------
__global__ __launch_bounds__(256) void rope_cache_k(float* __restrict__ cosb,
                                                    float* __restrict__ sinb) {
  int idx = blockIdx.x * 256 + threadIdx.x;   // 2048*32 total
  int p = idx >> 5, j = idx & 31;
  float ang = (float)p * __powf(10000.0f, -(float)j / 32.0f);
  float c = __cosf(ang), s = __sinf(ang);
  cosb[p * 64 + j] = c;  cosb[p * 64 + 32 + j] = c;
  sinb[p * 64 + j] = s;  sinb[p * 64 + 32 + j] = s;
}

// ---------------------------------------------------------------------------
// GEMM: C[M,N] = A[M,K] * W[N,K]^T. A, W bf16 in global; C bf16 or f32.
// Block tile 128x64, K-step 32, 8 waves x (32x32). Staging = async 16B copies.
// Requires M%128==0, N%64==0, K%32==0 (holds at every call site).
// ---------------------------------------------------------------------------
#define LDT 40   // padded LDS row stride in ushorts (80B, 16B-aligned)

template <bool OUT_BF16>
__global__ __launch_bounds__(256) void gemm_bf16_k(
    const unsigned short* __restrict__ A, int lda,
    const unsigned short* __restrict__ W, int ldw,
    void* __restrict__ Cv, int ldc, int K) {
  __shared__ __align__(16) unsigned short As[128 * LDT];
  __shared__ __align__(16) unsigned short Bs[64 * LDT];

  const int tid  = threadIdx.x;
  const int m0   = blockIdx.y * 128;
  const int n0   = blockIdx.x * 64;
  const int wid  = tid >> 5;
  const int lane = tid & 31;
  const int half = lane >> 4;
  const int r    = lane & 15;
  const int mw   = (wid >> 1) * 32;
  const int nw   = (wid & 1) * 32;

  v8f acc[2][2] = {};

  for (int k0 = 0; k0 < K; k0 += 32) {
    __syncthreads();
    // A tile: 128 rows x 32 cols = 512 x 16B chunks (2 per thread)
#pragma unroll
    for (int i = 0; i < 2; ++i) {
      int idx = i * 256 + tid;
      int row = idx >> 2, c = idx & 3;
      const unsigned short* src = A + (size_t)(m0 + row) * lda + k0 + c * 8;
      cp16_g2l(src, &As[row * LDT + c * 8]);
      if (k0 + 32 < K) __builtin_prefetch(src + 32, 0, 1);
    }
    // W tile: 64 rows x 32 cols = 256 x 16B chunks (1 per thread)
    {
      int row = tid >> 2, c = tid & 3;
      cp16_g2l(W + (size_t)(n0 + row) * ldw + k0 + c * 8, &Bs[row * LDT + c * 8]);
    }
    wait_cp();
    __syncthreads();

    v16bf a0 = load_frag(&As[(mw + r) * LDT], half);
    v16bf a1 = load_frag(&As[(mw + 16 + r) * LDT], half);
    v16bf b0 = load_frag(&Bs[(nw + r) * LDT], half);
    v16bf b1 = load_frag(&Bs[(nw + 16 + r) * LDT], half);

    acc[0][0] = wmma_bf16(a0, b0, acc[0][0]);
    acc[0][1] = wmma_bf16(a0, b1, acc[0][1]);
    acc[1][0] = wmma_bf16(a1, b0, acc[1][0]);
    acc[1][1] = wmma_bf16(a1, b1, acc[1][1]);
  }

#pragma unroll
  for (int im = 0; im < 2; ++im)
#pragma unroll
    for (int in = 0; in < 2; ++in)
#pragma unroll
      for (int e = 0; e < 8; ++e) {
        size_t row = (size_t)(m0 + mw + im * 16 + half * 8 + e);
        int    col = n0 + nw + in * 16 + r;
        if (OUT_BF16)
          ((unsigned short*)Cv)[row * ldc + col] = f2bf(acc[im][in][e]);
        else
          ((float*)Cv)[row * ldc + col] = acc[im][in][e];
      }
}

// ---------------------------------------------------------------------------
// In-place RMSNorm on bf16 rows (f32 accumulate). stride allows normalizing
// the 512-wide prefix of the 576-wide kv rows.
// ---------------------------------------------------------------------------
__global__ __launch_bounds__(256) void rmsnorm_bf16_k(unsigned short* __restrict__ x,
                                                      const float* __restrict__ w,
                                                      int dim, int stride) {
  const int row = blockIdx.x, tid = threadIdx.x;
  unsigned short* xr = x + (size_t)row * stride;
  float ss = 0.f;
  for (int i = tid; i < dim; i += 256) { float v = bf2f(xr[i]); ss += v * v; }
  __shared__ float red[256];
  red[tid] = ss;
  __syncthreads();
  for (int off = 128; off; off >>= 1) {
    if (tid < off) red[tid] += red[tid + off];
    __syncthreads();
  }
  float sc = rsqrtf(red[0] / (float)dim + 1e-6f);
  for (int i = tid; i < dim; i += 256) xr[i] = f2bf(w[i] * bf2f(xr[i]) * sc);
}

// ---------------------------------------------------------------------------
// Prep: RoPE q_pe/k_pe, fold softmax scale into q, emit bf16 q/k (S,H,192)
// and v (S,H,128). One block per sequence position. All inputs bf16.
// ---------------------------------------------------------------------------
__global__ __launch_bounds__(256) void prep_k(
    const unsigned short* __restrict__ qbuf,    // (S, 3072) = (S,H,192)
    const unsigned short* __restrict__ kvbuf,   // (S, 576), last 64 = k_pe
    const unsigned short* __restrict__ kvexp,   // (S, 4096) = (S,H,256)
    const int*   __restrict__ pos,
    const float* __restrict__ cosb, const float* __restrict__ sinb,
    unsigned short* __restrict__ qcat,
    unsigned short* __restrict__ kcat,
    unsigned short* __restrict__ vcat) {
  const int s = blockIdx.x, tid = threadIdx.x;
  int p = pos[s];
  if (p < 0) p = 0;
  if (p > 2047) p = 2047;
  const float* cr = cosb + p * 64;
  const float* sr = sinb + p * 64;

  __shared__ float kpe[64];
  if (tid < 64) {
    int dd = tid;
    float x = bf2f(kvbuf[(size_t)s * 576 + 512 + dd]);
    float rot = (dd < 32) ? -bf2f(kvbuf[(size_t)s * 576 + 512 + dd + 32])
                          :  bf2f(kvbuf[(size_t)s * 576 + 512 + dd - 32]);
    kpe[dd] = x * cr[dd] + rot * sr[dd];
  }
  __syncthreads();

  const float SCALE = 0.07216878364870323f;   // 192^-0.5

  for (int i = tid; i < 16 * 192; i += 256) {
    int h = i / 192, d = i % 192;
    const unsigned short* qr = qbuf + (size_t)s * 3072 + h * 192;
    float qv;
    if (d < 128) {
      qv = bf2f(qr[d]);
    } else {
      int dd = d - 128;
      float x = bf2f(qr[128 + dd]);
      float rot = (dd < 32) ? -bf2f(qr[128 + dd + 32]) : bf2f(qr[128 + dd - 32]);
      qv = x * cr[dd] + rot * sr[dd];
    }
    qcat[((size_t)s * 16 + h) * 192 + d] = f2bf(qv * SCALE);

    float kv = (d < 128) ? bf2f(kvexp[(size_t)s * 4096 + h * 256 + d])
                         : kpe[d - 128];
    kcat[((size_t)s * 16 + h) * 192 + d] = f2bf(kv);
  }
  for (int i = tid; i < 16 * 128; i += 256) {
    int h = i / 128, dv = i & 127;
    vcat[((size_t)s * 16 + h) * 128 + dv] = kvexp[(size_t)s * 4096 + h * 256 + 128 + dv];
  }
}

// ---------------------------------------------------------------------------
// Causal flash attention: one block per (head, 128 query rows), 8 waves x 16
// rows. Q fragments in registers for the whole pass; per 32-key block: K tile
// async-copied to LDS, V^T sync-transposed, 12 WMMAs (QK), online softmax via
// shfl, P re-layout through per-wave LDS (s_wait_dscnt), 8 WMMAs (PV).
// Output written directly as bf16 for the o-projection.
// ---------------------------------------------------------------------------
__global__ __launch_bounds__(256) void mla_attn_k(
    const unsigned short* __restrict__ qcat,   // (S,H,192) bf16 pre-scaled
    const unsigned short* __restrict__ kcat,   // (S,H,192) bf16
    const unsigned short* __restrict__ vcat,   // (S,H,128) bf16
    unsigned short* __restrict__ attnb) {      // (S, H*128) bf16
  __shared__ __align__(16) unsigned short Ks[32 * 200];
  __shared__ __align__(16) unsigned short Vt[128 * 40];
  __shared__ __align__(16) unsigned short Ps[8 * 16 * 40];

  const int tid  = threadIdx.x;
  const int h    = blockIdx.y;
  const int q0   = blockIdx.x * 128;
  const int wid  = tid >> 5;
  const int lane = tid & 31;
  const int half = lane >> 4;
  const int r    = lane & 15;

  const unsigned short* qrow = qcat + ((size_t)(q0 + wid * 16 + r) * 16 + h) * 192;
  v16bf aq[6];
#pragma unroll
  for (int kk = 0; kk < 6; ++kk) aq[kk] = load_frag(qrow + kk * 32, half);

  v8f o[8] = {};
  float mr[8], lr[8];
#pragma unroll
  for (int e = 0; e < 8; ++e) { mr[e] = -1e30f; lr[e] = 0.f; }

  const int nblocks = q0 / 32 + 4;   // causal: keys <= q0+127
  for (int jb = 0; jb < nblocks; ++jb) {
    const int j0 = jb * 32;
    __syncthreads();
    // K tile: 32 x 192 bf16 = 768 x 16B chunks (3 per thread), async
#pragma unroll
    for (int i = 0; i < 3; ++i) {
      int idx = i * 256 + tid;
      int row = idx / 24, c = idx % 24;
      cp16_g2l(kcat + ((size_t)(j0 + row) * 16 + h) * 192 + c * 8,
               &Ks[row * 200 + c * 8]);
    }
    // V^T tile: transpose scatter (can't be a 16B async copy)
    for (int i = tid; i < 32 * 128; i += 256) {
      int row = i >> 7, dv = i & 127;
      Vt[dv * 40 + row] = vcat[((size_t)(j0 + row) * 16 + h) * 128 + dv];
    }
    wait_cp();
    __syncthreads();

    v8f s0 = {}, s1 = {};
#pragma unroll
    for (int kk = 0; kk < 6; ++kk) {
      v16bf b0 = load_frag(&Ks[r * 200 + kk * 32], half);
      v16bf b1 = load_frag(&Ks[(16 + r) * 200 + kk * 32], half);
      s0 = wmma_bf16(aq[kk], b0, s0);
      s1 = wmma_bf16(aq[kk], b1, s1);
    }

    const int mbase = q0 + wid * 16 + half * 8;
    const int nc0 = j0 + r, nc1 = j0 + 16 + r;
#pragma unroll
    for (int e = 0; e < 8; ++e) {
      const int mg = mbase + e;
      float x0 = (nc0 <= mg) ? s0[e] : -1e30f;
      float x1 = (nc1 <= mg) ? s1[e] : -1e30f;
      float mx = fmaxf(x0, x1);
#pragma unroll
      for (int msk = 1; msk < 16; msk <<= 1)
        mx = fmaxf(mx, __shfl_xor(mx, msk, 32));
      float mnew  = fmaxf(mr[e], mx);        // finite after jb==0 (key 0 valid)
      float alpha = __expf(mr[e] - mnew);
      float p0 = __expf(x0 - mnew);          // masked -> exp(-huge) == 0
      float p1 = __expf(x1 - mnew);
      float rs = p0 + p1;
#pragma unroll
      for (int msk = 1; msk < 16; msk <<= 1)
        rs += __shfl_xor(rs, msk, 32);
      lr[e] = lr[e] * alpha + rs;
      mr[e] = mnew;
#pragma unroll
      for (int t = 0; t < 8; ++t) o[t][e] *= alpha;
      Ps[wid * 640 + (half * 8 + e) * 40 + r]      = f2bf(p0);
      Ps[wid * 640 + (half * 8 + e) * 40 + 16 + r] = f2bf(p1);
    }

    asm volatile("s_wait_dscnt 0" ::: "memory");  // P stores -> fragment loads

    v16bf ap = load_frag(&Ps[wid * 640 + r * 40], half);
#pragma unroll
    for (int t = 0; t < 8; ++t) {
      v16bf bv = load_frag(&Vt[(t * 16 + r) * 40], half);
      o[t] = wmma_bf16(ap, bv, o[t]);
    }
  }

  const int srow = q0 + wid * 16 + half * 8;
#pragma unroll
  for (int e = 0; e < 8; ++e) {
    float inv = 1.0f / fmaxf(lr[e], 1e-20f);
#pragma unroll
    for (int t = 0; t < 8; ++t)
      attnb[(size_t)(srow + e) * 2048 + h * 128 + t * 16 + r] = f2bf(o[t][e] * inv);
  }
}

// ---------------------------------------------------------------------------
// Host side
// ---------------------------------------------------------------------------
extern "C" void kernel_launch(void* const* d_in, const int* in_sizes, int n_in,
                              void* d_out, int out_size, void* d_ws, size_t ws_size,
                              hipStream_t stream) {
  (void)in_sizes; (void)n_in; (void)out_size; (void)ws_size;

  const float* hidden  = (const float*)d_in[0];   // (1,2048,2048)
  const int*   pos     = (const int*)  d_in[1];   // (1,2048)
  const float* q_a_w   = (const float*)d_in[2];   // (1536,2048)
  const float* q_a_ln  = (const float*)d_in[3];   // (1536,)
  const float* q_b_w   = (const float*)d_in[4];   // (3072,1536)
  const float* kv_a_w  = (const float*)d_in[5];   // (576,2048)
  const float* kv_a_ln = (const float*)d_in[6];   // (512,)
  const float* kv_b_w  = (const float*)d_in[7];   // (4096,512)
  const float* o_w     = (const float*)d_in[8];   // (2048,2048)
  float* out = (float*)d_out;                     // (1,2048,2048) f32

  size_t off = 0;
  auto take = [&](size_t bytes) -> void* {
    void* p = (char*)d_ws + off;
    off += (bytes + 255) & ~(size_t)255;
    return p;
  };
  float* cosb = (float*)take((size_t)2048 * 64 * 4);
  float* sinb = (float*)take((size_t)2048 * 64 * 4);
  unsigned short* hidb  = (unsigned short*)take((size_t)2048 * 2048 * 2);
  unsigned short* qawb  = (unsigned short*)take((size_t)1536 * 2048 * 2);
  unsigned short* qbwb  = (unsigned short*)take((size_t)3072 * 1536 * 2);
  unsigned short* kvawb = (unsigned short*)take((size_t)576 * 2048 * 2);
  unsigned short* kvbwb = (unsigned short*)take((size_t)4096 * 512 * 2);
  unsigned short* owb   = (unsigned short*)take((size_t)2048 * 2048 * 2);
  unsigned short* qa    = (unsigned short*)take((size_t)2048 * 1536 * 2);
  unsigned short* qbuf  = (unsigned short*)take((size_t)2048 * 3072 * 2);
  unsigned short* kvbuf = (unsigned short*)take((size_t)2048 * 576 * 2);
  unsigned short* kvexp = (unsigned short*)take((size_t)2048 * 4096 * 2);
  unsigned short* qcat  = (unsigned short*)take((size_t)2048 * 16 * 192 * 2);
  unsigned short* kcat  = (unsigned short*)take((size_t)2048 * 16 * 192 * 2);
  unsigned short* vcat  = (unsigned short*)take((size_t)2048 * 16 * 128 * 2);
  unsigned short* attnb = (unsigned short*)take((size_t)2048 * 2048 * 2);

  dim3 blk(256);

  rope_cache_k<<<256, blk, 0, stream>>>(cosb, sinb);

  // one-time bf16 conversions (weights + hidden)
  cvt_bf16_k<<<4096, blk, 0, stream>>>(hidden, hidb, 2048 * 2048);
  cvt_bf16_k<<<4096, blk, 0, stream>>>(q_a_w, qawb, 1536 * 2048);
  cvt_bf16_k<<<4096, blk, 0, stream>>>(q_b_w, qbwb, 3072 * 1536);
  cvt_bf16_k<<<4096, blk, 0, stream>>>(kv_a_w, kvawb, 576 * 2048);
  cvt_bf16_k<<<4096, blk, 0, stream>>>(kv_b_w, kvbwb, 4096 * 512);
  cvt_bf16_k<<<4096, blk, 0, stream>>>(o_w, owb, 2048 * 2048);

  // q_a = hidden @ q_a_w^T
  gemm_bf16_k<true><<<dim3(1536 / 64, 2048 / 128), blk, 0, stream>>>(
      hidb, 2048, qawb, 2048, qa, 1536, 2048);
  rmsnorm_bf16_k<<<2048, blk, 0, stream>>>(qa, q_a_ln, 1536, 1536);
  // q = q_a_norm @ q_b_w^T
  gemm_bf16_k<true><<<dim3(3072 / 64, 2048 / 128), blk, 0, stream>>>(
      qa, 1536, qbwb, 1536, qbuf, 3072, 1536);

  // kv = hidden @ kv_a_w^T
  gemm_bf16_k<true><<<dim3(576 / 64, 2048 / 128), blk, 0, stream>>>(
      hidb, 2048, kvawb, 2048, kvbuf, 576, 2048);
  rmsnorm_bf16_k<<<2048, blk, 0, stream>>>(kvbuf, kv_a_ln, 512, 576);
  // kv_exp = ckv @ kv_b_w^T
  gemm_bf16_k<true><<<dim3(4096 / 64, 2048 / 128), blk, 0, stream>>>(
      kvbuf, 576, kvbwb, 512, kvexp, 4096, 512);

  // RoPE + concat + softmax scale -> bf16 q/k/v
  prep_k<<<2048, blk, 0, stream>>>(qbuf, kvbuf, kvexp, pos, cosb, sinb,
                                   qcat, kcat, vcat);

  // causal flash attention (bf16 out)
  mla_attn_k<<<dim3(2048 / 128, 16), blk, 0, stream>>>(qcat, kcat, vcat, attnb);

  // out = attn @ o_w^T (f32 out)
  gemm_bf16_k<false><<<dim3(2048 / 64, 2048 / 128), blk, 0, stream>>>(
      attnb, 2048, owb, 2048, out, 2048, 2048);
}
